// MultiBranchGraphMamba_53111565583001
// MI455X (gfx1250) — matrix operand order repaired and checked by reading, not compiled, not run on hardware
//
#include <hip/hip_runtime.h>
#include <hip/hip_bf16.h>
#include <stddef.h>

// ---------------------------------------------------------------------------
// Problem constants (match reference)
// ---------------------------------------------------------------------------
#define BB 4
#define TT 16
#define NN 2000
#define FF 64
#define DD 64
#define EE 32000
#define NUM_OUT 8
#define BT (BB * TT)          // 64 graphs
#define ROWS (BT * NN)        // 128000 rows for the GEMM

typedef __attribute__((ext_vector_type(16))) __bf16 v16bf;
typedef __attribute__((ext_vector_type(8)))  __bf16 v8bf;
typedef __attribute__((ext_vector_type(8)))  float  v8f;

__device__ __forceinline__ float sigmoidf_(float x) { return 1.0f / (1.0f + __expf(-x)); }
__device__ __forceinline__ float siluf_(float x)    { return x * sigmoidf_(x); }

// ---------------------------------------------------------------------------
// Stage 0: degree / norm / CSR-by-destination build
// ---------------------------------------------------------------------------
__global__ void k_init(float* deg, int* cnt, int* fill) {
  int n = blockIdx.x * blockDim.x + threadIdx.x;
  if (n < NN) { deg[n] = 1.0f; cnt[n] = 1; fill[n] = 0; }  // self-loop seeds
}

__global__ void k_count(const int* __restrict__ ei, float* deg, int* cnt) {
  int e = blockIdx.x * blockDim.x + threadIdx.x;
  if (e < EE) {
    int d = ei[EE + e];               // edge_index[1][e]
    atomicAdd(&deg[d], 1.0f);
    atomicAdd(&cnt[d], 1);
  }
}

__global__ void k_rsqrt(float* deg) {  // in-place: deg -> dinv
  int n = blockIdx.x * blockDim.x + threadIdx.x;
  if (n < NN) deg[n] = rsqrtf(deg[n]);
}

// Single-block Hillis-Steele scan over 2048 slots (N=2000 fits).
__global__ void k_scan(const int* __restrict__ cnt, int* __restrict__ offs) {
  __shared__ int s[2048];
  int t = threadIdx.x;  // blockDim.x == 1024
  for (int i = t; i < 2048; i += 1024) s[i] = (i < NN) ? cnt[i] : 0;
  __syncthreads();
  for (int off = 1; off < 2048; off <<= 1) {
    int i0 = t, i1 = t + 1024;
    int v0 = (i0 >= off) ? s[i0 - off] : 0;
    int v1 = (i1 >= off) ? s[i1 - off] : 0;
    __syncthreads();
    s[i0] += v0; s[i1] += v1;
    __syncthreads();
  }
  for (int i = t; i < NN; i += 1024) offs[i + 1] = s[i];
  if (t == 0) offs[0] = 0;
}

__global__ void k_fill(const int* __restrict__ ei, const float* __restrict__ dinv,
                       const int* __restrict__ offs, int* fill,
                       int* __restrict__ ssrc, float* __restrict__ snorm) {
  int e = blockIdx.x * blockDim.x + threadIdx.x;
  if (e < EE) {
    int s = ei[e], d = ei[EE + e];
    int p = offs[d] + atomicAdd(&fill[d], 1);
    ssrc[p]  = s;
    snorm[p] = dinv[s] * dinv[d];
  } else if (e < EE + NN) {
    int n = e - EE;                    // self-loop
    int p = offs[n] + atomicAdd(&fill[n], 1);
    ssrc[p]  = n;
    snorm[p] = dinv[n] * dinv[n];
  }
}

// ---------------------------------------------------------------------------
// x (f32) -> Hb (bf16), one pass.  4 floats per thread.
// ---------------------------------------------------------------------------
__global__ void k_cvt(const float* __restrict__ x, __bf16* __restrict__ Hb) {
  size_t i = ((size_t)blockIdx.x * blockDim.x + threadIdx.x) * 4;
  float4 v = *(const float4*)(x + i);
  __bf16* o = Hb + i;
  o[0] = (__bf16)v.x; o[1] = (__bf16)v.y; o[2] = (__bf16)v.z; o[3] = (__bf16)v.w;
}

// ---------------------------------------------------------------------------
// Pack W (64x64 f32 row-major) into per-lane WMMA B fragments (bf16, 8 KB).
// Layout: Wp[((ct*2 + frag)*32 + lane)*16 + j], frag = K-block (0: K<32, 1: K>=32)
//   lane: n = lane&15, half = lane>>4 ; element j -> K = frag*32 + (j<8 ? half*8 + j : 16 + half*8 + (j-8))
// Each lane's 16 bf16 (32 B) are contiguous -> one aligned vector load.
// ---------------------------------------------------------------------------
__global__ void k_packW(const float* __restrict__ W, __bf16* __restrict__ Wp) {
  for (int p = threadIdx.x; p < 4096; p += 256) {
    int j      = p & 15;
    int lane   = (p >> 4) & 31;
    int fragid = p >> 9;               // ct*2 + frag
    int ct = fragid >> 1, frag = fragid & 1;
    int n = lane & 15, half = lane >> 4;
    int k = frag * 32 + (j < 8 ? half * 8 + j : 16 + half * 8 + (j - 8));
    Wp[p] = (__bf16)W[k * 64 + ct * 16 + n];
  }
}

// ---------------------------------------------------------------------------
// Stage 1a: GEMM  Hw = Hb @ W  via v_wmma_f32_16x16x32_bf16.
//   One wave32 computes a 16x64 slab: 4 column tiles x (K=64 -> 2 WMMAs).
//   A: 4 aligned 16B bf16 loads/lane. B: 8 aligned 32B loads/lane from Wp.
//   No LDS, no per-element conversion. Grid exact -> EXEC all-ones at WMMA.
// ---------------------------------------------------------------------------
__global__ void __launch_bounds__(256)
k_gemm(const __bf16* __restrict__ Hb, const __bf16* __restrict__ Wp,
       float* __restrict__ Hw) {
  const int lane = threadIdx.x & 31;
  const int wave = blockIdx.x * 8 + (threadIdx.x >> 5);
  const int row0 = wave * 16;           // 1000 blocks * 8 waves * 16 = 128000
  const int m    = lane & 15;
  const int half = lane >> 4;

  // A fragments (16-bit A 16x32 layout): element runs K=half*8+{0..7} and
  // K=16+half*8+{0..7} are contiguous 16B chunks -> v8bf loads + concat.
  const __bf16* arow = Hb + (size_t)(row0 + m) * 64;
  v8bf a0lo = *(const v8bf*)(arow + half * 8);
  v8bf a0hi = *(const v8bf*)(arow + 16 + half * 8);
  v8bf a1lo = *(const v8bf*)(arow + 32 + half * 8);
  v8bf a1hi = *(const v8bf*)(arow + 48 + half * 8);
  v16bf a0 = __builtin_shufflevector(a0lo, a0hi, 0,1,2,3,4,5,6,7,8,9,10,11,12,13,14,15);
  v16bf a1 = __builtin_shufflevector(a1lo, a1hi, 0,1,2,3,4,5,6,7,8,9,10,11,12,13,14,15);

  v8f acc[4];
  #pragma unroll
  for (int ct = 0; ct < 4; ++ct) {
    v16bf b0 = *(const v16bf*)(Wp + ((size_t)(ct * 2 + 0) * 32 + lane) * 16);
    v16bf b1 = *(const v16bf*)(Wp + ((size_t)(ct * 2 + 1) * 32 + lane) * 16);
    v8f c = {};
    c = __builtin_amdgcn_wmma_f32_16x16x32_bf16(false, a0, false, b0,
                                                (short)0, c, false, false);
    c = __builtin_amdgcn_wmma_f32_16x16x32_bf16(false, a1, false, b1,
                                                (short)0, c, false, false);
    acc[ct] = c;
  }

  // C/D layout: VGPR j -> row (8*half + j), col = lane&15 within tile
  #pragma unroll
  for (int ct = 0; ct < 4; ++ct)
    #pragma unroll
    for (int j = 0; j < 8; ++j)
      Hw[(size_t)(row0 + half * 8 + j) * 64 + ct * 16 + m] = acc[ct][j];
}

// ---------------------------------------------------------------------------
// Stage 1b: CSR aggregation + bias + SiLU; f32 accumulate, bf16 output.
//   64 threads per (graph,node): thread owns one feature channel.
// ---------------------------------------------------------------------------
__global__ void __launch_bounds__(256)
k_agg(const float* __restrict__ Hw, const int* __restrict__ offs,
      const int* __restrict__ ssrc, const float* __restrict__ snorm,
      const float* __restrict__ bias, __bf16* __restrict__ Hout) {
  int gn = blockIdx.x * 4 + (threadIdx.x >> 6);   // 32000 blocks * 4 = 128000
  int f  = threadIdx.x & 63;
  int g  = gn / NN;
  int n  = gn - g * NN;
  int k0 = offs[n], k1 = offs[n + 1];
  const float* base = Hw + (size_t)g * NN * 64;
  float acc = 0.0f;
  for (int k = k0; k < k1; ++k)
    acc += snorm[k] * base[(size_t)ssrc[k] * 64 + f];
  Hout[(size_t)gn * 64 + f] = (__bf16)siluf_(acc + bias[f]);
}

// ---------------------------------------------------------------------------
// Stage 2: mean pool over nodes -> pooled[g][d]  (f32 accumulate)
// ---------------------------------------------------------------------------
__global__ void k_pool(const __bf16* __restrict__ H, float* __restrict__ pooled) {
  int g = blockIdx.x, d = threadIdx.x;   // 64 blocks x 64 threads
  const __bf16* base = H + (size_t)g * NN * 64 + d;
  float s = 0.0f;
  for (int n = 0; n < NN; ++n) s += (float)base[(size_t)n * 64];
  pooled[g * 64 + d] = s * (1.0f / (float)NN);
}

// ---------------------------------------------------------------------------
// Stage 3: single-layer LSTM over T=16, batch B=4 (one block, 1024 threads)
// ---------------------------------------------------------------------------
__global__ void __launch_bounds__(1024)
k_lstm(const float* __restrict__ pooled,
       const float* __restrict__ Wih, const float* __restrict__ Whh,
       const float* __restrict__ bih, const float* __restrict__ bhh,
       float* __restrict__ hT) {
  __shared__ float hS[BB][DD], cS[BB][DD], gS[BB][4 * DD];
  int t = threadIdx.x;
  int b = t >> 8, r = t & 255;          // gate-row worker mapping
  if (t < BB * DD) { hS[t >> 6][t & 63] = 0.0f; cS[t >> 6][t & 63] = 0.0f; }
  __syncthreads();
  for (int step = 0; step < TT; ++step) {
    const float* xt = pooled + (size_t)(b * TT + step) * DD;
    float acc = bih[r] + bhh[r];
    const float* wi = Wih + (size_t)r * DD;
    const float* wh = Whh + (size_t)r * DD;
    #pragma unroll 4
    for (int k = 0; k < DD; ++k) acc += xt[k] * wi[k] + hS[b][k] * wh[k];
    gS[b][r] = acc;
    __syncthreads();
    if (t < BB * DD) {
      int bb = t >> 6, d = t & 63;
      float i = sigmoidf_(gS[bb][d]);
      float f = sigmoidf_(gS[bb][64 + d]);
      float gg = tanhf(gS[bb][128 + d]);
      float o = sigmoidf_(gS[bb][192 + d]);
      float c = f * cS[bb][d] + i * gg;
      cS[bb][d] = c;
      hS[bb][d] = o * tanhf(c);
    }
    __syncthreads();
  }
  if (t < BB * DD) hT[t] = hS[t >> 6][t & 63];
}

// ---------------------------------------------------------------------------
// Stage 4: MLP head (one block, 512 threads)
// ---------------------------------------------------------------------------
__global__ void __launch_bounds__(512)
k_head(const float* __restrict__ hT,
       const float* __restrict__ Wp1, const float* __restrict__ bp1,
       const float* __restrict__ Wp2, const float* __restrict__ bp2,
       const float* __restrict__ Wp3, const float* __restrict__ bp3,
       float* __restrict__ out) {
  __shared__ float z1[BB][2 * DD], z2[BB][DD];
  int t = threadIdx.x;
  { // z1 = silu(hT @ Wp1 + bp1)  : (4,64)@(64,128)
    int b = t >> 7, j = t & 127;
    float acc = bp1[j];
    for (int k = 0; k < DD; ++k) acc += hT[b * DD + k] * Wp1[k * 128 + j];
    z1[b][j] = siluf_(acc);
  }
  __syncthreads();
  if (t < BB * DD) { // z2 = silu(z1 @ Wp2 + bp2) : (4,128)@(128,64)
    int b = t >> 6, j = t & 63;
    float acc = bp2[j];
    for (int k = 0; k < 2 * DD; ++k) acc += z1[b][k] * Wp2[k * 64 + j];
    z2[b][j] = siluf_(acc);
  }
  __syncthreads();
  if (t < BB * NUM_OUT) { // out = z2 @ Wp3 + bp3 : (4,64)@(64,8)
    int b = t >> 3, j = t & 7;
    float acc = bp3[j];
    for (int k = 0; k < DD; ++k) acc += z2[b][k] * Wp3[k * NUM_OUT + j];
    out[b * NUM_OUT + j] = acc;
  }
}

// ---------------------------------------------------------------------------
// Host launcher
// ---------------------------------------------------------------------------
extern "C" void kernel_launch(void* const* d_in, const int* in_sizes, int n_in,
                              void* d_out, int out_size, void* d_ws, size_t ws_size,
                              hipStream_t stream) {
  const float* x   = (const float*)d_in[0];
  const int*   ei  = (const int*)d_in[1];
  const float* W1  = (const float*)d_in[2];  const float* b1  = (const float*)d_in[3];
  const float* W2  = (const float*)d_in[4];  const float* b2  = (const float*)d_in[5];
  const float* W3  = (const float*)d_in[6];  const float* b3  = (const float*)d_in[7];
  const float* Wih = (const float*)d_in[8];  const float* Whh = (const float*)d_in[9];
  const float* bih = (const float*)d_in[10]; const float* bhh = (const float*)d_in[11];
  const float* Wp1 = (const float*)d_in[12]; const float* bp1 = (const float*)d_in[13];
  const float* Wp2 = (const float*)d_in[14]; const float* bp2 = (const float*)d_in[15];
  const float* Wp3 = (const float*)d_in[16]; const float* bp3 = (const float*)d_in[17];
  float* out = (float*)d_out;

  // Workspace carve-up (~50 MB)
  char* ws = (char*)d_ws;
  const size_t HW_BYTES = (size_t)ROWS * 64 * sizeof(float);   // 32.768 MB
  const size_t HB_BYTES = (size_t)ROWS * 64 * sizeof(__bf16);  // 16.384 MB
  float*  Hw   = (float*)ws;
  __bf16* Hb   = (__bf16*)(ws + HW_BYTES);
  char*   sm   = ws + HW_BYTES + HB_BYTES;
  float*  dinv  = (float*)(sm);                   // 2000 f (deg -> dinv in place)
  int*    cnt   = (int*)  (sm + 8192);            // 2000 i
  int*    offs  = (int*)  (sm + 16384);           // 2001 i
  int*    fill  = (int*)  (sm + 24576);           // 2000 i
  int*    ssrc  = (int*)  (sm + 32768);           // 34000 i
  float*  snorm = (float*)(sm + 32768 + 136192);  // 34000 f
  float*  pooled= (float*)(sm + 32768 + 2 * 136192);          // 64*64 f
  float*  hT    = (float*)(sm + 32768 + 2 * 136192 + 16384);  // 4*64 f
  __bf16* Wpk   = (__bf16*)(sm + 32768 + 2 * 136192 + 20480); // 4096 bf16

  // --- graph preprocessing (per call, no persistent state) ---
  k_init  <<<(NN + 255) / 256, 256, 0, stream>>>(dinv, cnt, fill);
  k_count <<<(EE + 255) / 256, 256, 0, stream>>>(ei, dinv, cnt);
  k_rsqrt <<<(NN + 255) / 256, 256, 0, stream>>>(dinv);
  k_scan  <<<1, 1024, 0, stream>>>(cnt, offs);
  k_fill  <<<(EE + NN + 255) / 256, 256, 0, stream>>>(ei, dinv, offs, fill, ssrc, snorm);

  // --- activations to bf16 once ---
  k_cvt<<<(ROWS * 64 / 4) / 256, 256, 0, stream>>>(x, Hb);   // 8000 blocks

  // --- 3x (pack W -> WMMA GEMM -> CSR aggregate + bias + SiLU) ---
  const int gemm_blocks = ROWS / 16 / 8;   // 1000 (exact; EXEC all-ones for WMMA)
  const int agg_blocks  = ROWS / 4;        // 32000
  k_packW<<<1, 256, 0, stream>>>(W1, Wpk);
  k_gemm<<<gemm_blocks, 256, 0, stream>>>(Hb, Wpk, Hw);
  k_agg <<<agg_blocks, 256, 0, stream>>>(Hw, offs, ssrc, snorm, b1, Hb);
  k_packW<<<1, 256, 0, stream>>>(W2, Wpk);
  k_gemm<<<gemm_blocks, 256, 0, stream>>>(Hb, Wpk, Hw);
  k_agg <<<agg_blocks, 256, 0, stream>>>(Hw, offs, ssrc, snorm, b2, Hb);
  k_packW<<<1, 256, 0, stream>>>(W3, Wpk);
  k_gemm<<<gemm_blocks, 256, 0, stream>>>(Hb, Wpk, Hw);
  k_agg <<<agg_blocks, 256, 0, stream>>>(Hw, offs, ssrc, snorm, b3, Hb);

  // --- pool, LSTM, head ---
  k_pool<<<BT, 64, 0, stream>>>(Hb, pooled);
  k_lstm<<<1, 1024, 0, stream>>>(pooled, Wih, Whh, bih, bhh, hT);
  k_head<<<1, 512, 0, stream>>>(hT, Wp1, bp1, Wp2, bp2, Wp3, bp3, out);
}
